// MultiheadAttention_1984274891094
// MI455X (gfx1250) — matrix operand order repaired and checked by reading, not compiled, not run on hardware
//
#include <hip/hip_runtime.h>
#include <hip/hip_bf16.h>

typedef __attribute__((ext_vector_type(16))) _Float16 v16h;
typedef __attribute__((ext_vector_type(8)))  _Float16 v8h;
typedef __attribute__((ext_vector_type(8)))  float    v8f;

#define HID   1024
#define NH    16
#define HD    64
#define SEQ   2048
#define BATCH 2
#define MTOK  (BATCH * SEQ)   // 4096

// ---------------------------------------------------------------------------
// CDNA5 async global->LDS copy (ASYNCcnt path), portable via inline asm.
// lds_off: wave-relative LDS byte address (low 32 bits of generic pointer).
// ---------------------------------------------------------------------------
static __device__ __forceinline__ void async_ld_b128(unsigned lds_off, const void* g) {
  unsigned long long ga = (unsigned long long)(uintptr_t)g;
  asm volatile("global_load_async_to_lds_b128 %0, %1, off"
               :: "v"(lds_off), "v"(ga) : "memory");
}
static __device__ __forceinline__ void wait_async0() {
  asm volatile("s_wait_asynccnt 0x0" ::: "memory");
}
static __device__ __forceinline__ unsigned lds_off32(const void* p) {
  return (unsigned)(uintptr_t)p;   // LDS aperture: addr[31:0] = wave-relative offset
}

// ---------------------------------------------------------------------------
// WMMA fragment helpers (16x16x32 f16 -> f32)
// ---------------------------------------------------------------------------
static __device__ __forceinline__ v16h cat8(v8h lo, v8h hi) {
  return __builtin_shufflevector(lo, hi, 0, 1, 2, 3, 4, 5, 6, 7,
                                          8, 9, 10, 11, 12, 13, 14, 15);
}

// A fragment: 16x32 f16 tile, row-major [M][ldk] (halves).
// lanes 0-15 -> M=lane, K chunks [0..7],[16..23]; lanes 16-31 -> K [8..15],[24..31].
static __device__ __forceinline__ v16h load_frag_a(const _Float16* tile, int ldk) {
  const int lane = threadIdx.x & 31;
  const _Float16* p = tile + (lane & 15) * ldk + ((lane >> 4) << 3);
  v8h lo = *(const v8h*)(p);
  v8h hi = *(const v8h*)(p + 16);
  return cat8(lo, hi);
}

// B fragment: 32x16 (KxN) from row-major [N][ldk] tile.
// lanes 0-15 -> N=lane, K chunks [0..7],[16..23]; lanes 16-31 -> N=lane-16, K [8..15],[24..31].
static __device__ __forceinline__ v16h load_frag_b(const _Float16* tile, int ldk) {
  const int lane = threadIdx.x & 31;
  const _Float16* p = tile + (lane & 15) * ldk + ((lane >> 4) << 3);
  v8h lo = *(const v8h*)(p);
  v8h hi = *(const v8h*)(p + 16);
  return cat8(lo, hi);
}

static __device__ __forceinline__ v8f wmma16(v16h a, v16h b, v8f c) {
  return __builtin_amdgcn_wmma_f32_16x16x32_f16(false, a, false, b, (short)0, c, false, false);
}

// ---------------------------------------------------------------------------
// f32 -> f16 conversion
// ---------------------------------------------------------------------------
__global__ __launch_bounds__(256) void mha_cvt_f16(const float* __restrict__ in,
                                                   _Float16* __restrict__ out, int n) {
  int i = blockIdx.x * 256 + threadIdx.x;
  if (i < n) out[i] = (_Float16)in[i];
}

// ---------------------------------------------------------------------------
// Tiled GEMM: C[M,N] = A[M,K] * B[N,K]^T + bias[N]
// 256 threads / 8 waves; 128x128 tile; K-step 32; async-double-buffered LDS.
// MODE 0: q/k head-major f16 [b,h,s,d]; v stored TRANSPOSED [b,h,d,s].
// MODE 1: f32 row-major output.
// ---------------------------------------------------------------------------
template <int MODE>
__global__ __launch_bounds__(256) void mha_gemm128(const _Float16* __restrict__ A,
                                                   const _Float16* __restrict__ Bm,
                                                   const float* __restrict__ bias,
                                                   float* __restrict__ outF,
                                                   _Float16* __restrict__ outQKV,
                                                   int K) {
  __shared__ _Float16 As[2][128 * 32];
  __shared__ _Float16 Bs[2][128 * 32];

  const int tid  = threadIdx.x;
  const int lane = tid & 31;
  const int w    = tid >> 5;
  const int wm   = w & 3;
  const int wn   = w >> 2;
  const int mBase = blockIdx.y * 128;
  const int nBase = blockIdx.x * 128;

  // issue async loads of one K-slab into buffer `buf`
  auto issue = [&](int k0, int buf) {
#pragma unroll
    for (int t = 0; t < 2; ++t) {
      int i   = tid + t * 256;   // 0..511
      int row = i >> 2;
      int seg = i & 3;
      async_ld_b128(lds_off32(&As[buf][0]) + i * 16,
                    A + (size_t)(mBase + row) * K + k0 + seg * 8);
      async_ld_b128(lds_off32(&Bs[buf][0]) + i * 16,
                    Bm + (size_t)(nBase + row) * K + k0 + seg * 8);
    }
  };

  v8f acc[2][4];
#pragma unroll
  for (int i = 0; i < 2; ++i)
#pragma unroll
    for (int j = 0; j < 4; ++j) {
      v8f z = {};
      acc[i][j] = z;
    }

  issue(0, 0);
  wait_async0();
  __syncthreads();

  for (int k0 = 0; k0 < K; k0 += 32) {
    const int buf = (k0 >> 5) & 1;
    if (k0 + 32 < K) issue(k0 + 32, buf ^ 1);   // prefetch next slab

    v16h a0 = load_frag_a(&As[buf][(wm * 32 + 0)  * 32], 32);
    v16h a1 = load_frag_a(&As[buf][(wm * 32 + 16) * 32], 32);
#pragma unroll
    for (int j = 0; j < 4; ++j) {
      v16h bf = load_frag_b(&Bs[buf][(wn * 64 + j * 16) * 32], 32);
      acc[0][j] = wmma16(a0, bf, acc[0][j]);
      acc[1][j] = wmma16(a1, bf, acc[1][j]);
    }

    wait_async0();
    __syncthreads();
  }

  const int hi8 = (lane >> 4) << 3;
  const int col = lane & 15;
#pragma unroll
  for (int i = 0; i < 2; ++i)
#pragma unroll
    for (int j = 0; j < 4; ++j)
#pragma unroll
      for (int r = 0; r < 8; ++r) {
        int mg = mBase + wm * 32 + i * 16 + hi8 + r;
        int ng = nBase + wn * 64 + j * 16 + col;
        float val = acc[i][j][r] + bias[ng];
        if (MODE == 0) {
          int part = ng >> 10;            // 0=q 1=k 2=v
          int rem  = ng & 1023;
          int hh   = rem >> 6;
          int dd   = rem & 63;
          int bb   = mg >> 11;
          int ss   = mg & 2047;
          size_t bh = (size_t)(bb * NH + hh);
          if (part == 2) {
            // V stored transposed: [b,h,d,s]
            outQKV[(size_t)2 * (MTOK * HID) + (bh * HD + dd) * SEQ + ss] = (_Float16)val;
          } else {
            outQKV[(size_t)part * (MTOK * HID) + (bh * SEQ + ss) * HD + dd] = (_Float16)val;
          }
        } else {
          outF[(size_t)mg * HID + ng] = val;
        }
      }
}

// ---------------------------------------------------------------------------
// Fused causal flash attention, async-double-buffered K / V^T tiles.
// Grid: BATCH*NH*(SEQ/128) blocks of 256 threads (8 waves).
// Dynamic LDS: 2*(128x64) K + 2*(64x128) V^T + 8*(16x128) P = 96 KB.
// Reference semantics: mask to -10000 BEFORE the 1/sqrt(64) scale.
// ---------------------------------------------------------------------------
__global__ __launch_bounds__(256) void mha_flash_attn(const _Float16* __restrict__ Q,
                                                      const _Float16* __restrict__ Kg,
                                                      const _Float16* __restrict__ Vt_g,
                                                      _Float16* __restrict__ ctx) {
  extern __shared__ _Float16 smem[];
  _Float16* KtBuf = smem;                    // 2 * 128*64
  _Float16* VtBuf = KtBuf + 2 * 128 * HD;    // 2 * 64*128
  _Float16* Pb    = VtBuf + 2 * HD * 128;    // 8 * 16*128

  const int tid  = threadIdx.x;
  const int lane = tid & 31;
  const int w    = tid >> 5;
  const int qt   = blockIdx.x & 15;          // SEQ/128 = 16
  const int bh   = blockIdx.x >> 4;          // 0..31
  const int b    = bh >> 4;
  const int h    = bh & 15;

  const _Float16* qb  = Q    + (size_t)bh * SEQ * HD;
  const _Float16* kb  = Kg   + (size_t)bh * SEQ * HD;
  const _Float16* vtb = Vt_g + (size_t)bh * HD * SEQ;   // [d][s]

  const int hi8  = (lane >> 4) << 3;
  const int col0 = lane & 15;
  const int qRow = qt * 128 + w * 16;

  // issue async loads for key-tile kt into buffer `buf`
  auto issueTiles = [&](int kt, int buf) {
    _Float16* Kt = KtBuf + buf * 128 * HD;
    _Float16* Vt = VtBuf + buf * HD * 128;
#pragma unroll
    for (int t = 0; t < 4; ++t) {
      int i = tid + t * 256;                 // 0..1023
      // K tile: row-major [key][d], contiguous slab
      async_ld_b128(lds_off32(Kt) + i * 16,
                    kb + (size_t)kt * 128 * HD + i * 8);
      // V^T tile: [d][key], row d stride SEQ in global
      int d   = i >> 4;
      int seg = i & 15;
      async_ld_b128(lds_off32(Vt) + (d * 128 + seg * 8) * 2,
                    vtb + (size_t)d * SEQ + kt * 128 + seg * 8);
    }
  };

  // Resident Q fragments (16 rows x 64 head dims = 2 K-steps)
  v16h aq0 = load_frag_a(qb + (size_t)qRow * HD, HD);
  v16h aq1 = load_frag_a(qb + (size_t)qRow * HD + 32, HD);

  v8f o[4];
#pragma unroll
  for (int j = 0; j < 4; ++j) {
    v8f z = {};
    o[j] = z;
  }
  float mrow[8], lrow[8];
#pragma unroll
  for (int r = 0; r < 8; ++r) {
    mrow[r] = -1e30f;
    lrow[r] = 0.0f;
  }

  _Float16* Pw = Pb + w * 16 * 128;

  issueTiles(0, 0);
  wait_async0();
  __syncthreads();

  for (int kt = 0; kt <= qt; ++kt) {
    const int buf = kt & 1;
    if (kt < qt) issueTiles(kt + 1, buf ^ 1);   // prefetch next key tile

    const _Float16* Kt = KtBuf + buf * 128 * HD;
    const _Float16* Vt = VtBuf + buf * HD * 128;

    // ---- scores: S(16x128) = Q(16x64) * K^T ----
    v8f s[8];
#pragma unroll
    for (int j = 0; j < 8; ++j) {
      v8f z = {};
      s[j] = z;
      v16h bk0 = load_frag_b(Kt + (j * 16) * HD, HD);
      v16h bk1 = load_frag_b(Kt + (j * 16) * HD + 32, HD);
      s[j] = wmma16(aq0, bk0, s[j]);
      s[j] = wmma16(aq1, bk1, s[j]);
    }

    const bool last = (kt == qt);
    const int rowLocalBase = w * 16 + hi8;

    // ---- mask (before scale, per reference) + scale + row max ----
    float rm[8];
#pragma unroll
    for (int r = 0; r < 8; ++r) {
      float mx = -1e30f;
#pragma unroll
      for (int j = 0; j < 8; ++j) {
        float vsc = s[j][r];
        int colLocal = j * 16 + col0;
        int rowLocal = rowLocalBase + r;
        bool msk = last && (colLocal > rowLocal);
        vsc = msk ? -1250.0f : vsc * 0.125f;   // (-10000)/8 for masked
        s[j][r] = vsc;
        mx = fmaxf(mx, vsc);
      }
      mx = fmaxf(mx, __shfl_xor(mx, 1, 32));
      mx = fmaxf(mx, __shfl_xor(mx, 2, 32));
      mx = fmaxf(mx, __shfl_xor(mx, 4, 32));
      mx = fmaxf(mx, __shfl_xor(mx, 8, 32));
      rm[r] = mx;
    }

    // ---- online softmax update ----
#pragma unroll
    for (int r = 0; r < 8; ++r) {
      float mnew = fmaxf(mrow[r], rm[r]);
      float sf   = __expf(mrow[r] - mnew);
      mrow[r] = mnew;
      float rs = 0.0f;
#pragma unroll
      for (int j = 0; j < 8; ++j) {
        float p = __expf(s[j][r] - mnew);
        s[j][r] = p;
        rs += p;
      }
      rs += __shfl_xor(rs, 1, 32);
      rs += __shfl_xor(rs, 2, 32);
      rs += __shfl_xor(rs, 4, 32);
      rs += __shfl_xor(rs, 8, 32);
      lrow[r] = lrow[r] * sf + rs;
#pragma unroll
      for (int j2 = 0; j2 < 4; ++j2) o[j2][r] *= sf;
    }

    // ---- P -> LDS (C-layout to A-fragment layout round trip) ----
#pragma unroll
    for (int j = 0; j < 8; ++j)
#pragma unroll
      for (int r = 0; r < 8; ++r)
        Pw[(hi8 + r) * 128 + j * 16 + col0] = (_Float16)s[j][r];
    asm volatile("s_wait_dscnt 0x0" ::: "memory");   // wave-local LDS visibility

    // ---- O += P(16x128) * V(128x64) ----
#pragma unroll
    for (int ks = 0; ks < 4; ++ks) {
      v16h ap = load_frag_a(Pw + ks * 32, 128);
#pragma unroll
      for (int j2 = 0; j2 < 4; ++j2) {
        v16h bv = load_frag_b(Vt + (j2 * 16) * 128 + ks * 32, 128);
        o[j2] = wmma16(ap, bv, o[j2]);
      }
    }

    wait_async0();      // prefetched tiles landed (this wave)
    __syncthreads();    // all waves done reading current + next tiles ready
  }

  // ---- normalize and write context back to [B,S,H] (f16 for final GEMM) ----
#pragma unroll
  for (int j2 = 0; j2 < 4; ++j2)
#pragma unroll
    for (int r = 0; r < 8; ++r) {
      int srow = qt * 128 + w * 16 + hi8 + r;
      int dd   = j2 * 16 + col0;
      float val = o[j2][r] / lrow[r];
      ctx[((size_t)(b * SEQ + srow)) * HID + h * HD + dd] = (_Float16)val;
    }
}

// ---------------------------------------------------------------------------
// Host-side launch
// ---------------------------------------------------------------------------
extern "C" void kernel_launch(void* const* d_in, const int* in_sizes, int n_in,
                              void* d_out, int out_size, void* d_ws, size_t ws_size,
                              hipStream_t stream) {
  const float* x      = (const float*)d_in[0];  // [2,2048,1024]
  const float* W_attn = (const float*)d_in[1];  // [3072,1024]
  const float* b_attn = (const float*)d_in[2];  // [3072]
  const float* W_out  = (const float*)d_in[3];  // [1024,1024]
  const float* b_out  = (const float*)d_in[4];  // [1024]
  float* out = (float*)d_out;                   // [2,2048,1024]

  _Float16* ws   = (_Float16*)d_ws;
  _Float16* xh   = ws;                          // 4096*1024
  _Float16* wah  = xh  + (size_t)MTOK * HID;    // 3072*1024
  _Float16* woh  = wah + (size_t)3 * HID * HID; // 1024*1024
  _Float16* qkvh = woh + (size_t)HID * HID;     // 3 * 4096*1024
  _Float16* qh   = qkvh;                        // [b,h,s,d]
  _Float16* kh   = qh + (size_t)MTOK * HID;     // [b,h,s,d]
  _Float16* vth  = kh + (size_t)MTOK * HID;     // [b,h,d,s]  (transposed!)
  _Float16* ctxh = vth + (size_t)MTOK * HID;    // 4096*1024

  {
    int n1 = MTOK * HID;
    int n2 = 3 * HID * HID;
    int n3 = HID * HID;
    mha_cvt_f16<<<(n1 + 255) / 256, 256, 0, stream>>>(x, xh, n1);
    mha_cvt_f16<<<(n2 + 255) / 256, 256, 0, stream>>>(W_attn, wah, n2);
    mha_cvt_f16<<<(n3 + 255) / 256, 256, 0, stream>>>(W_out, woh, n3);
  }

  {
    dim3 grid(3 * HID / 128, MTOK / 128);   // (24, 32)
    mha_gemm128<0><<<grid, 256, 0, stream>>>(xh, wah, b_attn, nullptr, qkvh, HID);
  }

  {
    int blocks = BATCH * NH * (SEQ / 128);  // 512
    size_t lds_bytes = (size_t)(2 * 128 * HD + 2 * HD * 128 + 8 * 16 * 128) * sizeof(_Float16); // 96 KB
    mha_flash_attn<<<blocks, 256, lds_bytes, stream>>>(qh, kh, vth, ctxh);
  }

  {
    dim3 grid(HID / 128, MTOK / 128);       // (8, 32)
    mha_gemm128<1><<<grid, 256, 0, stream>>>(ctxh, woh, b_out, out, nullptr, HID);
  }
}
